// PointNet_34995393528525
// MI455X (gfx1250) — matrix operand order repaired and checked by reading, not compiled, not run on hardware
//
#include <hip/hip_runtime.h>

// ---------------------------------------------------------------------------
// CDNA5 (gfx1250) PointNet GNN: bf16 WMMA data path, wave32, per-wave tiles.
// ---------------------------------------------------------------------------

typedef __bf16 bf16_t;
typedef __attribute__((ext_vector_type(16))) __bf16    v16bf;
typedef __attribute__((ext_vector_type(8)))  float     v8f;
typedef __attribute__((ext_vector_type(8)))  unsigned  v8u;
typedef __attribute__((ext_vector_type(4)))  unsigned  v4u;

// Wave-local LDS fence: DS ops are in-order per wave, but cross-lane
// producer->consumer through LDS needs an explicit dscnt wait (compiler only
// tracks per-lane deps). gfx1250 split-counter wait (CDNA5 ISA).
__device__ inline void wave_lds_fence() {
  asm volatile("s_wait_dscnt 0" ::: "memory");
}

__device__ inline void atom_add_f32(float* p, float v) {
  unsafeAtomicAdd(p, v);   // hardware global_atomic_add_f32
}

// Single-instruction ReLU: med3(v, 0, +inf) -> v_med3_num_f32 (fmaxf costs
// two v_max_num_f32 due to NaN canonicalization).
__device__ inline float relu1(float v) {
  return __builtin_amdgcn_fmed3f(v, 0.0f, __builtin_inff());
}

__device__ inline unsigned pack_bf16(float a, float b) {
  union { bf16_t h[2]; unsigned u; } t;
  t.h[0] = (bf16_t)a;
  t.h[1] = (bf16_t)b;
  return t.u;
}

// --- WMMA fragment loaders (layouts per CDNA5 ISA §7.12.2) ------
// A fragment: 16x32 bf16 tile, row-major with leading dim ldk.
// lane<16 holds row M=lane, K pairs {0..7},{16..23}; lane>=16 same row,
// K pairs {8..15},{24..31}. Both 4-dword groups are 16B aligned ->
// two ds_load_b128 per fragment.
__device__ inline v16bf frag_a(const bf16_t* t, int ldk, int kbase, int lane) {
  const int row = lane & 15, half = lane >> 4;
  const v4u* p = (const v4u*)(t + row * ldk + kbase);  // 16B units
  const v4u lo = p[half];       // dwords half*4 .. +3   (K group 0)
  const v4u hi = p[2 + half];   // dwords 8+half*4 .. +3 (K group 1)
  v8u u;
#pragma unroll
  for (int q = 0; q < 4; ++q) { u[q] = lo[q]; u[q + 4] = hi[q]; }
  return __builtin_bit_cast(v16bf, u);
}

// B fragment: 32x16 bf16 from row-major K x N weight (leading dim ldn).
// lane&15 = N column; elements are K {8h..8h+7},{16+8h..16+8h+7} (same as A).
__device__ inline v16bf frag_b(const bf16_t* B, int ldn, int kbase, int nbase,
                               int lane) {
  const int half = lane >> 4;
  const int ncol = nbase + (lane & 15);
  v16bf v;
#pragma unroll
  for (int s = 0; s < 16; ++s) {
    const int k = kbase + (s < 8 ? 8 * half + s : 16 + 8 * half + (s - 8));
    v[s] = B[(long)k * ldn + ncol];
  }
  return v;
}

// Fold the bias into the WMMA C operand: D = A*B + C with C[m][n] = bias[n].
__device__ inline void init_bias(v8f acc[4], const float* __restrict__ b,
                                 int lane) {
  const int n0 = lane & 15;
#pragma unroll
  for (int nb = 0; nb < 4; ++nb) {
    const float bv = b[nb * 16 + n0];
#pragma unroll
    for (int q = 0; q < 8; ++q) acc[nb][q] = bv;
  }
}

// One wave computes D(16x64) = A(16x(32*KT)) @ B((32*KT)x64) + C, bf16/f32.
template <int KT>
__device__ inline void wave_gemm(const bf16_t* A, const bf16_t* B, v8f acc[4],
                                 int lane) {
#pragma unroll
  for (int kt = 0; kt < KT; ++kt) {
    const v16bf a = frag_a(A, KT * 32, kt * 32, lane);
#pragma unroll
    for (int nb = 0; nb < 4; ++nb) {
      const v16bf b = frag_b(B, 64, kt * 32, nb * 16, lane);
      acc[nb] = __builtin_amdgcn_wmma_f32_16x16x32_bf16(
          false, a, false, b, (short)0, acc[nb], false, false);
    }
  }
}

// ---------------------------------------------------------------------------
// Utility kernels
// ---------------------------------------------------------------------------
__global__ void zero_kernel(float* __restrict__ p, long n) {
  long i = (long)blockIdx.x * blockDim.x + threadIdx.x;
  const long stride = (long)gridDim.x * blockDim.x;
  for (; i < n; i += stride) p[i] = 0.0f;
}

// f32 (K x Nn) -> bf16 (Kp x Nn), rows K..Kp-1 zero-filled (exact K padding).
__global__ void cvt_pad_kernel(const float* __restrict__ s,
                               bf16_t* __restrict__ d, int K, int Kp, int Nn) {
  const long tot = (long)Kp * Nn;
  long i = (long)blockIdx.x * blockDim.x + threadIdx.x;
  if (i >= tot) return;
  const int k = (int)(i / Nn), nn = (int)(i % Nn);
  const float v = (k < K) ? s[(long)k * Nn + nn] : 0.0f;
  d[i] = (bf16_t)v;
}

// ---------------------------------------------------------------------------
// node_lin: h = relu(x @ W1 + b1) @ W2 + b2        (x: N x 16, h: N x 64 bf16)
// ---------------------------------------------------------------------------
__global__ __launch_bounds__(256) void node_mlp_kernel(
    const float* __restrict__ x, const bf16_t* __restrict__ W1p,  // 32 x 64
    const float* __restrict__ b1, const bf16_t* __restrict__ W2,  // 64 x 64
    const float* __restrict__ b2, bf16_t* __restrict__ hout, int n) {
  __shared__ __align__(16) bf16_t sA[8][16 * 32];
  __shared__ __align__(16) bf16_t sH[8][16 * 64];
  const int wave = threadIdx.x >> 5, lane = threadIdx.x & 31;
  const long tile = (long)blockIdx.x * 8 + wave;
  const long r0 = tile * 16;
  if (r0 >= n) return;
  bf16_t* A = sA[wave];
  const int r = lane >> 1, hs = lane & 1;   // 2 lanes per row
  const long i = r0 + r;
  const bool valid = i < n;
  const float* xr = x + (valid ? i : 0) * 16;
  unsigned* Ad = (unsigned*)(A + r * 32);   // 16 dwords per row
  if (hs == 0) {
#pragma unroll
    for (int j = 0; j < 8; ++j)
      Ad[j] = valid ? pack_bf16(xr[2 * j], xr[2 * j + 1]) : 0u;
  } else {
#pragma unroll
    for (int j = 8; j < 16; ++j) Ad[j] = 0u;     // K-pad 16..31
  }
  wave_lds_fence();

  v8f acc[4]; init_bias(acc, b1, lane);
  wave_gemm<1>(A, W1p, acc, lane);

  bf16_t* Ht = sH[wave];
  const int n0 = lane & 15, half = lane >> 4;
#pragma unroll
  for (int nb = 0; nb < 4; ++nb)
#pragma unroll
    for (int q = 0; q < 8; ++q) {
      const int m = q + half * 8, nn = nb * 16 + n0;
      Ht[m * 64 + nn] = (bf16_t)relu1(acc[nb][q]);
    }
  wave_lds_fence();

  v8f acc2[4]; init_bias(acc2, b2, lane);
  wave_gemm<2>(Ht, W2, acc2, lane);
#pragma unroll
  for (int q = 0; q < 8; ++q) {
    const int m = q + half * 8;
    const long row = r0 + m;
    if (row < n) {
#pragma unroll
      for (int nb = 0; nb < 4; ++nb)
        hout[row * 64 + nb * 16 + n0] = (bf16_t)acc2[nb][q];  // no relu here
    }
  }
}

// ---------------------------------------------------------------------------
// Edge message + scatter-add:
//   m = relu([h[dst], h[src], pos[src]] @ W1 + b1) @ W2 + b2 ; agg[dst] += m
// K = 131 padded to 160 (W1 pre-padded with zero rows).
// ---------------------------------------------------------------------------
__global__ __launch_bounds__(256) void edge_message_kernel(
    const bf16_t* __restrict__ h, const float* __restrict__ pos,
    const int* __restrict__ srcI, const int* __restrict__ dstI,
    const bf16_t* __restrict__ W1, const float* __restrict__ b1,  // 160 x 64
    const bf16_t* __restrict__ W2, const float* __restrict__ b2,  // 64 x 64
    float* __restrict__ agg, int nE) {
  __shared__ __align__(16) bf16_t sA[8][16 * 160];
  __shared__ __align__(16) bf16_t sM[8][16 * 64];
  __shared__ int sD[8][16];
  const int wave = threadIdx.x >> 5, lane = threadIdx.x & 31;
  const long tile = (long)blockIdx.x * 8 + wave;
  const long e0 = tile * 16;
  if (e0 >= nE) return;
  // Weights are reused by every tile: pull them toward the caches early
  // (lowers to global_prefetch_b8).
  __builtin_prefetch(W1, 0, 1);
  __builtin_prefetch(W2, 0, 1);
  bf16_t* A = sA[wave];

  // Gather: 2 lanes per edge row. Row = [h_dst(64) | h_src(64) | pos(3) | 0s].
  const int r = lane >> 1, hs = lane & 1;
  const long e = e0 + r;
  const bool valid = e < nE;
  int di = 0, si = 0;
  if (valid) { di = dstI[e]; si = srcI[e]; }
  if (hs == 0) sD[wave][r] = valid ? di : -1;
  const v4u* hd4 = (const v4u*)(h + (long)di * 64);   // 8 x 16B per row
  const v4u* hs4 = (const v4u*)(h + (long)si * 64);
  v4u* Arow4 = (v4u*)(A + r * 160);                   // 20 x 16B per row
  const v4u z4 = {0u, 0u, 0u, 0u};
#pragma unroll
  for (int j = 0; j < 4; ++j) {
    const int c4 = hs * 4 + j;
    Arow4[c4]     = valid ? hd4[c4] : z4;             // cols   0..63
    Arow4[8 + c4] = valid ? hs4[c4] : z4;             // cols  64..127
  }
  if (hs == 1) {                                      // cols 128..159
    unsigned* Ad = (unsigned*)(A + r * 160);
    const float* pp = pos + (long)si * 3;
    Ad[64] = valid ? pack_bf16(pp[0], pp[1]) : 0u;
    Ad[65] = valid ? pack_bf16(pp[2], 0.0f) : 0u;
#pragma unroll
    for (int j = 66; j < 80; ++j) Ad[j] = 0u;
  }
  wave_lds_fence();

  v8f acc[4]; init_bias(acc, b1, lane);
  wave_gemm<5>(A, W1, acc, lane);                     // K=160

  bf16_t* M = sM[wave];
  const int n0 = lane & 15, half = lane >> 4;
#pragma unroll
  for (int nb = 0; nb < 4; ++nb)
#pragma unroll
    for (int q = 0; q < 8; ++q) {
      const int m = q + half * 8, nn = nb * 16 + n0;
      M[m * 64 + nn] = (bf16_t)relu1(acc[nb][q]);
    }
  wave_lds_fence();

  v8f acc2[4]; init_bias(acc2, b2, lane);
  wave_gemm<2>(M, W2, acc2, lane);                    // K=64

  // scatter-add into agg[dst]
#pragma unroll
  for (int q = 0; q < 8; ++q) {
    const int m = q + half * 8;
    const int d = sD[wave][m];
    if (d >= 0) {
#pragma unroll
      for (int nb = 0; nb < 4; ++nb)
        atom_add_f32(&agg[(long)d * 64 + nb * 16 + n0], acc2[nb][q]);
    }
  }
}

// ---------------------------------------------------------------------------
// global_nn: h = relu( relu(agg @ W1 + b1) @ W2 + b2 )   (agg f32 -> h bf16)
// ---------------------------------------------------------------------------
__global__ __launch_bounds__(256) void global_mlp_kernel(
    const float* __restrict__ agg, const bf16_t* __restrict__ W1,
    const float* __restrict__ b1, const bf16_t* __restrict__ W2,
    const float* __restrict__ b2, bf16_t* __restrict__ hout, int n) {
  __shared__ __align__(16) bf16_t sA[8][16 * 64];
  __shared__ __align__(16) bf16_t sH[8][16 * 64];
  const int wave = threadIdx.x >> 5, lane = threadIdx.x & 31;
  const long tile = (long)blockIdx.x * 8 + wave;
  const long r0 = tile * 16;
  if (r0 >= n) return;
  bf16_t* A = sA[wave];
  const int r = lane >> 1, hs = lane & 1;
  const long i = r0 + r;
  const bool valid = i < n;
  const float* ar = agg + (valid ? i : 0) * 64;
  unsigned* Ad = (unsigned*)(A + r * 64);   // 32 dwords per row
#pragma unroll
  for (int j = 0; j < 16; ++j) {
    const int c = hs * 16 + j;
    Ad[c] = valid ? pack_bf16(ar[2 * c], ar[2 * c + 1]) : 0u;
  }
  wave_lds_fence();

  v8f acc[4]; init_bias(acc, b1, lane);
  wave_gemm<2>(A, W1, acc, lane);

  bf16_t* Ht = sH[wave];
  const int n0 = lane & 15, half = lane >> 4;
#pragma unroll
  for (int nb = 0; nb < 4; ++nb)
#pragma unroll
    for (int q = 0; q < 8; ++q) {
      const int m = q + half * 8, nn = nb * 16 + n0;
      Ht[m * 64 + nn] = (bf16_t)relu1(acc[nb][q]);
    }
  wave_lds_fence();

  v8f acc2[4]; init_bias(acc2, b2, lane);
  wave_gemm<2>(Ht, W2, acc2, lane);
#pragma unroll
  for (int q = 0; q < 8; ++q) {
    const int m = q + half * 8;
    const long row = r0 + m;
    if (row < n) {
#pragma unroll
      for (int nb = 0; nb < 4; ++nb)
        hout[row * 64 + nb * 16 + n0] = (bf16_t)relu1(acc2[nb][q]);
    }
  }
}

// ---------------------------------------------------------------------------
// Head: t = relu(h @ lin1_W + b) ; out = t @ lin2_W + b ; readout per graph.
// ---------------------------------------------------------------------------
__global__ __launch_bounds__(256) void head_kernel(
    const bf16_t* __restrict__ h, const int* __restrict__ batch,
    const float* __restrict__ l1W, const float* __restrict__ l1b,
    const float* __restrict__ l2W, const float* __restrict__ l2b,
    float* __restrict__ out, int n) {
  __shared__ float sW1[64 * 32];
  __shared__ float sW2[32 * 8];
  __shared__ float sb1[32];
  __shared__ float sb2[8];
  for (int i = threadIdx.x; i < 64 * 32; i += blockDim.x) sW1[i] = l1W[i];
  for (int i = threadIdx.x; i < 32 * 8; i += blockDim.x) sW2[i] = l2W[i];
  if (threadIdx.x < 32) sb1[threadIdx.x] = l1b[threadIdx.x];
  if (threadIdx.x < 8) sb2[threadIdx.x] = l2b[threadIdx.x];
  __syncthreads();
  const long i = (long)blockIdx.x * blockDim.x + threadIdx.x;
  if (i >= n) return;
  float hr[64];
#pragma unroll
  for (int j = 0; j < 64; ++j) hr[j] = (float)h[i * 64 + j];
  float o[8];
#pragma unroll
  for (int c = 0; c < 8; ++c) o[c] = sb2[c];
  for (int k = 0; k < 32; ++k) {
    float t = sb1[k];
#pragma unroll
    for (int j = 0; j < 64; ++j) t += hr[j] * sW1[j * 32 + k];
    t = relu1(t);
#pragma unroll
    for (int c = 0; c < 8; ++c) o[c] += t * sW2[k * 8 + c];
  }
  const int g = batch[i];
#pragma unroll
  for (int c = 0; c < 8; ++c) atom_add_f32(&out[(long)g * 8 + c], o[c]);
}

// ---------------------------------------------------------------------------
// Host launch
// ---------------------------------------------------------------------------
extern "C" void kernel_launch(void* const* d_in, const int* in_sizes, int n_in,
                              void* d_out, int out_size, void* d_ws,
                              size_t ws_size, hipStream_t stream) {
  const int N = in_sizes[0] / 16;   // F = 16
  const int E = in_sizes[2] / 2;
  const int G = out_size / 8;
  const int L = in_sizes[9] / 64;   // local_b1: (L, 64)

  const float* x    = (const float*)d_in[0];
  const float* pos  = (const float*)d_in[1];
  const int*   ei   = (const int*)d_in[2];
  const int*   batch= (const int*)d_in[3];
  const float* nW1  = (const float*)d_in[4];
  const float* nb1  = (const float*)d_in[5];
  const float* nW2  = (const float*)d_in[6];
  const float* nb2  = (const float*)d_in[7];
  const float* lW1  = (const float*)d_in[8];
  const float* lb1  = (const float*)d_in[9];
  const float* lW2  = (const float*)d_in[10];
  const float* lb2  = (const float*)d_in[11];
  const float* gW1  = (const float*)d_in[12];
  const float* gb1  = (const float*)d_in[13];
  const float* gW2  = (const float*)d_in[14];
  const float* gb2  = (const float*)d_in[15];
  const float* l1W  = (const float*)d_in[16];
  const float* l1b  = (const float*)d_in[17];
  const float* l2W  = (const float*)d_in[18];
  const float* l2b  = (const float*)d_in[19];

  // Workspace layout: h (bf16 N x 64) | agg (f32 N x 64) | bf16 weights.
  char* ws = (char*)d_ws;
  bf16_t* h   = (bf16_t*)ws;
  float*  agg = (float*)(ws + ((size_t)N * 64 * 2 + 255) / 256 * 256);
  bf16_t* wb  = (bf16_t*)((char*)agg + ((size_t)N * 64 * 4 + 255) / 256 * 256);
  bf16_t* nW1p = wb;  wb += 32 * 64;
  bf16_t* nW2b = wb;  wb += 64 * 64;
  bf16_t* lW1p[8]; bf16_t* lW2b[8]; bf16_t* gW1b[8]; bf16_t* gW2b[8];
  for (int l = 0; l < L; ++l) {
    lW1p[l] = wb; wb += 160 * 64;
    lW2b[l] = wb; wb += 64 * 64;
    gW1b[l] = wb; wb += 64 * 64;
    gW2b[l] = wb; wb += 64 * 64;
  }

  auto cvt = [&](const float* s, bf16_t* d, int K, int Kp, int Nn) {
    const long tot = (long)Kp * Nn;
    cvt_pad_kernel<<<(int)((tot + 255) / 256), 256, 0, stream>>>(s, d, K, Kp, Nn);
  };
  cvt(nW1, nW1p, 16, 32, 64);
  cvt(nW2, nW2b, 64, 64, 64);
  for (int l = 0; l < L; ++l) {
    cvt(lW1 + (long)l * 131 * 64, lW1p[l], 131, 160, 64);
    cvt(lW2 + (long)l * 64 * 64,  lW2b[l], 64, 64, 64);
    cvt(gW1 + (long)l * 64 * 64,  gW1b[l], 64, 64, 64);
    cvt(gW2 + (long)l * 64 * 64,  gW2b[l], 64, 64, 64);
  }

  const int ntiles = (N + 15) / 16;
  const int nblocks = (ntiles + 7) / 8;
  node_mlp_kernel<<<nblocks, 256, 0, stream>>>(x, nW1p, nb1, nW2b, nb2, h, N);

  const int etiles = (E + 15) / 16;
  const int eblocks = (etiles + 7) / 8;
  for (int l = 0; l < L; ++l) {
    zero_kernel<<<2048, 256, 0, stream>>>(agg, (long)N * 64);
    edge_message_kernel<<<eblocks, 256, 0, stream>>>(
        h, pos, ei /*src*/, ei + E /*dst*/, lW1p[l], lb1 + (long)l * 64,
        lW2b[l], lb2 + (long)l * 64, agg, E);
    global_mlp_kernel<<<nblocks, 256, 0, stream>>>(
        agg, gW1b[l], gb1 + (long)l * 64, gW2b[l], gb2 + (long)l * 64, h, N);
  }

  zero_kernel<<<4, 256, 0, stream>>>((float*)d_out, (long)G * 8);
  head_kernel<<<(N + 255) / 256, 256, 0, stream>>>(h, batch, l1W, l1b, l2W,
                                                   l2b, (float*)d_out, N);
}
